// F5TTS_19731079758535
// MI455X (gfx1250) — matrix operand order repaired and checked by reading, not compiled, not run on hardware
//
#include <hip/hip_runtime.h>
#include <hip/hip_bf16.h>
#include <math.h>

#define B_      4
#define T_      1024
#define TTEXT   256
#define DIM     1024
#define HEADS   16
#define HEAD_DIM 64
#define NMEL    100
#define TEXTDIM 512
#define FFI     2048
#define VOCAB   2545

typedef __attribute__((ext_vector_type(16))) __bf16 v16bf;
typedef __attribute__((ext_vector_type(8)))  float  v8f;

__device__ __forceinline__ float gelu_f(float x) { return 0.5f * x * (1.0f + erff(x * 0.70710678118f)); }
__device__ __forceinline__ float silu_f(float x) { return x / (1.0f + __expf(-x)); }
__device__ __forceinline__ float act_apply(float x, int act) {
    return act == 1 ? gelu_f(x) : (act == 2 ? silu_f(x) : x);
}

// ---------------------------------------------------------------------------
// Tiled bf16 WMMA GEMM: C[M,N] = act(A[M,K] @ B[K,N] + bias) + resid
// fp32 operands converted to bf16 in-register during LDS staging (weights are
// L2-resident on MI455X's 192MB L2, so no pre-conversion pass is needed).
// Block tile 128x128, KT=32, 8 waves (2x4), each wave: 4x2 of 16x16 tiles.
// Staging uses a vectorized fast path (b128 loads, packed cvt, vector LDS
// stores); the guarded scalar path only runs on K/N boundary tiles.
// ---------------------------------------------------------------------------
__global__ __launch_bounds__(256) void k_gemm_wmma(
    const float* __restrict__ A, int lda,
    const float* __restrict__ Bm, int ldb,
    const float* __restrict__ bias,
    const float* __restrict__ resid,
    float* __restrict__ C, int ldc,
    int M, int N, int K, int act)
{
    __shared__ __bf16 sA[128][32];   // [m][k]
    __shared__ __bf16 sBt[128][32];  // [n][k] (transposed stage)

    const int tid  = threadIdx.x;
    const int lane = tid & 31;
    const int wid  = tid >> 5;
    const int wm   = wid >> 2;      // 0..1  -> 64 rows each
    const int wn   = wid & 3;       // 0..3  -> 32 cols each
    const int l16  = lane & 15;
    const int half = lane >> 4;
    const int m0 = blockIdx.y * 128, n0 = blockIdx.x * 128;

    v8f acc[4][2];
#pragma unroll
    for (int i = 0; i < 4; ++i)
#pragma unroll
        for (int j = 0; j < 2; ++j) acc[i][j] = (v8f)0.0f;

    const int arow = tid >> 1, ac0 = (tid & 1) * 16; // A: 1 row x 16 k per thread
    const int bn   = (tid & 31) * 4;                 // B: 4k x 4n per thread
    const int bk   = (tid >> 5) * 4;

    for (int k0 = 0; k0 < K; k0 += 32) {
        const bool fullK = (k0 + 32 <= K);

        // ---- stage A tile (128x32) ----
        {
            const int gm = m0 + arow;
            const float* ap = A + (size_t)gm * lda + (k0 + ac0);
            if (fullK && (m0 + 128 <= M)) {
                if (k0 + 64 <= K) __builtin_prefetch(ap + 32, 0, 1);
                const float4 f0 = *(const float4*)(ap + 0);
                const float4 f1 = *(const float4*)(ap + 4);
                const float4 f2 = *(const float4*)(ap + 8);
                const float4 f3 = *(const float4*)(ap + 12);
                union { __bf16 h[16]; uint4 u[2]; } pk;
                pk.h[0]  = (__bf16)f0.x; pk.h[1]  = (__bf16)f0.y; pk.h[2]  = (__bf16)f0.z; pk.h[3]  = (__bf16)f0.w;
                pk.h[4]  = (__bf16)f1.x; pk.h[5]  = (__bf16)f1.y; pk.h[6]  = (__bf16)f1.z; pk.h[7]  = (__bf16)f1.w;
                pk.h[8]  = (__bf16)f2.x; pk.h[9]  = (__bf16)f2.y; pk.h[10] = (__bf16)f2.z; pk.h[11] = (__bf16)f2.w;
                pk.h[12] = (__bf16)f3.x; pk.h[13] = (__bf16)f3.y; pk.h[14] = (__bf16)f3.z; pk.h[15] = (__bf16)f3.w;
                *(uint4*)&sA[arow][ac0 + 0] = pk.u[0];
                *(uint4*)&sA[arow][ac0 + 8] = pk.u[1];
            } else {
                const bool mok = gm < M;
#pragma unroll
                for (int i = 0; i < 16; ++i) {
                    const int k = k0 + ac0 + i;
                    sA[arow][ac0 + i] = (__bf16)((mok && k < K) ? ap[i] : 0.0f);
                }
            }
        }
        // ---- stage B tile (32x128), transposed into sBt[n][k] ----
        {
            const float* bp = Bm + (size_t)(k0 + bk) * ldb + (n0 + bn);
            if (fullK && (n0 + 128 <= N)) {
                if (k0 + 64 <= K) __builtin_prefetch(bp + (size_t)32 * ldb, 0, 1);
                const float4 r0 = *(const float4*)(bp);
                const float4 r1 = *(const float4*)(bp + (size_t)ldb);
                const float4 r2 = *(const float4*)(bp + 2 * (size_t)ldb);
                const float4 r3 = *(const float4*)(bp + 3 * (size_t)ldb);
                union { __bf16 h[4]; uint2 u; } c0, c1, c2, c3;
                c0.h[0] = (__bf16)r0.x; c0.h[1] = (__bf16)r1.x; c0.h[2] = (__bf16)r2.x; c0.h[3] = (__bf16)r3.x;
                c1.h[0] = (__bf16)r0.y; c1.h[1] = (__bf16)r1.y; c1.h[2] = (__bf16)r2.y; c1.h[3] = (__bf16)r3.y;
                c2.h[0] = (__bf16)r0.z; c2.h[1] = (__bf16)r1.z; c2.h[2] = (__bf16)r2.z; c2.h[3] = (__bf16)r3.z;
                c3.h[0] = (__bf16)r0.w; c3.h[1] = (__bf16)r1.w; c3.h[2] = (__bf16)r2.w; c3.h[3] = (__bf16)r3.w;
                *(uint2*)&sBt[bn + 0][bk] = c0.u;
                *(uint2*)&sBt[bn + 1][bk] = c1.u;
                *(uint2*)&sBt[bn + 2][bk] = c2.u;
                *(uint2*)&sBt[bn + 3][bk] = c3.u;
            } else {
#pragma unroll
                for (int kk = 0; kk < 4; ++kk) {
                    const int gk = k0 + bk + kk;
                    const bool kok = gk < K;
#pragma unroll
                    for (int nn = 0; nn < 4; ++nn) {
                        const int n = n0 + bn + nn;
                        sBt[bn + nn][bk + kk] =
                            (__bf16)((kok && n < N) ? Bm[(size_t)gk * ldb + n] : 0.0f);
                    }
                }
            }
        }
        __syncthreads();

        // ---- fragments per ISA layouts ----
        v16bf af[4], bf[2];
#pragma unroll
        for (int tm = 0; tm < 4; ++tm) {
            // A 16x32 bf16: lanes0-15 hold K 0..7 & 16..23; lanes16-31 K 8..15 & 24..31
            const __bf16* p = &sA[wm * 64 + tm * 16 + l16][half * 8];
#pragma unroll
            for (int i = 0; i < 8; ++i) { af[tm][i] = p[i]; af[tm][8 + i] = p[16 + i]; }
        }
#pragma unroll
        for (int tn = 0; tn < 2; ++tn) {
            // B 32x16 bf16: lanes0-15 hold K 0..7 & 16..23; lanes16-31 K 8..15 & 24..31 (col = l16)
            const __bf16* p = &sBt[wn * 32 + tn * 16 + l16][half * 8];
#pragma unroll
            for (int i = 0; i < 8; ++i) { bf[tn][i] = p[i]; bf[tn][8 + i] = p[16 + i]; }
        }
#pragma unroll
        for (int tm = 0; tm < 4; ++tm)
#pragma unroll
            for (int tn = 0; tn < 2; ++tn)
                acc[tm][tn] = __builtin_amdgcn_wmma_f32_16x16x32_bf16(
                    false, af[tm], false, bf[tn], (short)0, acc[tm][tn], false, false);
        __syncthreads();
    }

    // ---- epilogue: C/D layout -> m = r + 8*half, n = l16 within tile ----
#pragma unroll
    for (int tm = 0; tm < 4; ++tm)
#pragma unroll
        for (int tn = 0; tn < 2; ++tn)
#pragma unroll
            for (int r = 0; r < 8; ++r) {
                const int m = m0 + wm * 64 + tm * 16 + half * 8 + r;
                const int n = n0 + wn * 32 + tn * 16 + l16;
                if (m < M && n < N) {
                    float v = acc[tm][tn][r];
                    if (bias)  v += bias[n];
                    v = act_apply(v, act);
                    if (resid) v += resid[(size_t)m * ldc + n];
                    C[(size_t)m * ldc + n] = v;
                }
            }
}

// ---------------------------------------------------------------------------
// Skinny GEMM for M=4 conditioning matmuls (time MLP, AdaLN cond)
// ---------------------------------------------------------------------------
__global__ void k_small_gemm(const float* __restrict__ A, const float* __restrict__ W,
                             const float* __restrict__ bias, float* __restrict__ out,
                             int M, int N, int K, int act)
{
    const int n = blockIdx.x * blockDim.x + threadIdx.x;
    const int m = blockIdx.y;
    if (n >= N || m >= M) return;
    float s = bias ? bias[n] : 0.0f;
    for (int k = 0; k < K; ++k) s += A[(size_t)m * K + k] * W[(size_t)k * N + n];
    out[(size_t)m * N + n] = act_apply(s, act);
}

// ---------------------------------------------------------------------------
// Row LayerNorm with affine
// ---------------------------------------------------------------------------
__global__ __launch_bounds__(256) void k_ln(const float* __restrict__ x,
                                            const float* __restrict__ g,
                                            const float* __restrict__ b,
                                            float* __restrict__ y, int C)
{
    const int row = blockIdx.x;
    const float* xr = x + (size_t)row * C;
    __shared__ float red[256];
    float s = 0.0f;
    for (int c = threadIdx.x; c < C; c += 256) s += xr[c];
    red[threadIdx.x] = s; __syncthreads();
    for (int o = 128; o > 0; o >>= 1) { if (threadIdx.x < o) red[threadIdx.x] += red[threadIdx.x + o]; __syncthreads(); }
    const float mean = red[0] / C; __syncthreads();
    float s2 = 0.0f;
    for (int c = threadIdx.x; c < C; c += 256) { const float d = xr[c] - mean; s2 += d * d; }
    red[threadIdx.x] = s2; __syncthreads();
    for (int o = 128; o > 0; o >>= 1) { if (threadIdx.x < o) red[threadIdx.x] += red[threadIdx.x + o]; __syncthreads(); }
    const float inv = rsqrtf(red[0] / C + 1e-5f);
    for (int c = threadIdx.x; c < C; c += 256) {
        const float v = (xr[c] - mean) * inv;
        y[(size_t)row * C + c] = g ? v * g[c] + b[c] : v;
    }
}

// AdaLN: y = LN(x) * (1 + sc[b,c]) + sh[b,c]; cond = [B, 2*DIM] (sc | sh)
__global__ __launch_bounds__(256) void k_adaln(const float* __restrict__ x,
                                               const float* __restrict__ cond,
                                               float* __restrict__ y)
{
    const int row = blockIdx.x;          // 0..B*T-1
    const int bb  = row / T_;
    const float* xr = x + (size_t)row * DIM;
    __shared__ float red[256];
    float s = 0.0f;
    for (int c = threadIdx.x; c < DIM; c += 256) s += xr[c];
    red[threadIdx.x] = s; __syncthreads();
    for (int o = 128; o > 0; o >>= 1) { if (threadIdx.x < o) red[threadIdx.x] += red[threadIdx.x + o]; __syncthreads(); }
    const float mean = red[0] / DIM; __syncthreads();
    float s2 = 0.0f;
    for (int c = threadIdx.x; c < DIM; c += 256) { const float d = xr[c] - mean; s2 += d * d; }
    red[threadIdx.x] = s2; __syncthreads();
    for (int o = 128; o > 0; o >>= 1) { if (threadIdx.x < o) red[threadIdx.x] += red[threadIdx.x + o]; __syncthreads(); }
    const float inv = rsqrtf(red[0] / DIM + 1e-5f);
    for (int c = threadIdx.x; c < DIM; c += 256) {
        const float v  = (xr[c] - mean) * inv;
        const float sc = cond[(size_t)bb * 2 * DIM + c];
        const float sh = cond[(size_t)bb * 2 * DIM + DIM + c];
        y[(size_t)row * DIM + c] = v * (1.0f + sc) + sh;
    }
}

// Depthwise conv over time (convnext, K=7 pad=3)
__global__ void k_dwconv7(const float* __restrict__ x, const float* __restrict__ w,
                          const float* __restrict__ b, float* __restrict__ y)
{
    const size_t idx = (size_t)blockIdx.x * blockDim.x + threadIdx.x;
    if (idx >= (size_t)B_ * T_ * DIM) return;
    const int c  = idx & (DIM - 1);
    const int t  = (idx >> 10) & (T_ - 1);
    const int bb = idx >> 20;
    float s = b[c];
#pragma unroll
    for (int j = 0; j < 7; ++j) {
        const int tt = t + j - 3;
        if (tt >= 0 && tt < T_) s += x[((size_t)(bb * T_ + tt)) * DIM + c] * w[c * 7 + j];
    }
    y[idx] = s;
}

// Text path: xf[b,t,:] += gelu(dwconv3(txt)+b) for t < TTEXT (pad handled by bound)
__global__ void k_txtconv_add(const float* __restrict__ txt, const float* __restrict__ w,
                              const float* __restrict__ b, float* __restrict__ xf)
{
    const size_t idx = (size_t)blockIdx.x * blockDim.x + threadIdx.x;
    if (idx >= (size_t)B_ * TTEXT * DIM) return;
    const int c  = idx & (DIM - 1);
    const int t  = (idx >> 10) & (TTEXT - 1);
    const int bb = idx >> 18;
    float s = b[c];
#pragma unroll
    for (int j = 0; j < 3; ++j) {
        const int tt = t + j - 1;
        if (tt >= 0 && tt < TTEXT) s += txt[((size_t)(bb * TTEXT + tt)) * DIM + c] * w[c * 3 + j];
    }
    xf[((size_t)(bb * T_ + t)) * DIM + c] += gelu_f(s);
}

__global__ void k_gather(const int* __restrict__ tok, const float* __restrict__ embed,
                         float* __restrict__ out)
{
    const size_t idx = (size_t)blockIdx.x * blockDim.x + threadIdx.x;
    if (idx >= (size_t)B_ * TTEXT * TEXTDIM) return;
    const int c  = idx & (TEXTDIM - 1);
    const int t  = (idx >> 9) & (TTEXT - 1);
    const int bb = idx >> 17;
    out[idx] = embed[(size_t)tok[bb * TTEXT + t] * TEXTDIM + c];
}

__global__ void k_timefreq(const float* __restrict__ timev, float* __restrict__ e)
{
    const int idx = blockIdx.x * blockDim.x + threadIdx.x;
    if (idx >= B_ * DIM) return;
    const int i = idx & (DIM - 1), bb = idx >> 10;
    const int half = DIM / 2;
    const int j = (i < half) ? i : i - half;
    const float f = __expf(-9.210340372f * (float)j / (float)(half - 1)); // ln(10000)
    const float a = timev[bb] * f;
    e[idx] = (i < half) ? __sinf(a) : __cosf(a);
}

// GRN: gx[b,c] = sqrt(sum_t h^2)
__global__ void k_grn_gx(const float* __restrict__ h, float* __restrict__ gx)
{
    const int c = blockIdx.x * blockDim.x + threadIdx.x;
    const int bb = blockIdx.y;
    if (c >= FFI) return;
    float s = 0.0f;
    for (int t = 0; t < T_; ++t) { const float v = h[((size_t)(bb * T_ + t)) * FFI + c]; s += v * v; }
    gx[bb * FFI + c] = sqrtf(s);
}
__global__ __launch_bounds__(256) void k_grn_mean(const float* __restrict__ gx, float* __restrict__ gm)
{
    const int bb = blockIdx.x;
    __shared__ float red[256];
    float s = 0.0f;
    for (int c = threadIdx.x; c < FFI; c += 256) s += gx[bb * FFI + c];
    red[threadIdx.x] = s; __syncthreads();
    for (int o = 128; o > 0; o >>= 1) { if (threadIdx.x < o) red[threadIdx.x] += red[threadIdx.x + o]; __syncthreads(); }
    if (threadIdx.x == 0) gm[bb] = red[0] / FFI;
}
__global__ void k_grn_apply(float* __restrict__ h, const float* __restrict__ gx,
                            const float* __restrict__ gm, const float* __restrict__ gg,
                            const float* __restrict__ gb)
{
    const size_t idx = (size_t)blockIdx.x * blockDim.x + threadIdx.x;
    if (idx >= (size_t)B_ * T_ * FFI) return;
    const int c  = idx & (FFI - 1);
    const int bb = idx >> 21;
    const float nx = gx[bb * FFI + c] / (gm[bb] + 1e-6f);
    const float h0 = h[idx];
    h[idx] = gg[c] * (h0 * nx) + gb[c] + h0;
}

// RoPE on PE head (head 0) of q and k, in-place in qkv [B,T,3*DIM]
__global__ void k_rope(float* __restrict__ qkv)
{
    const int idx = blockIdx.x * blockDim.x + threadIdx.x;
    if (idx >= B_ * T_ * 32) return;
    const int i = idx & 31, t = (idx >> 5) & (T_ - 1), bb = idx >> 15;
    const float invf = __powf(10000.0f, -(float)(2 * i) / 64.0f);
    const float a = (float)t * invf;
    const float c = __cosf(a), s = __sinf(a);
    const size_t base = ((size_t)(bb * T_ + t)) * (3 * DIM);
    float x1 = qkv[base + i], x2 = qkv[base + 32 + i];
    qkv[base + i]      = x1 * c - x2 * s;
    qkv[base + 32 + i] = x2 * c + x1 * s;
    float y1 = qkv[base + DIM + i], y2 = qkv[base + DIM + 32 + i];
    qkv[base + DIM + i]      = y1 * c - y2 * s;
    qkv[base + DIM + 32 + i] = y2 * c + y1 * s;
}

// Flash-style attention: one wave per query, online softmax (attention is ~17%
// of FLOPs -> VALU streaming kernel, no O(T^2) workspace).
__global__ __launch_bounds__(256) void k_attn(const float* __restrict__ qkv,
                                              const unsigned char* __restrict__ mask,
                                              float* __restrict__ out)
{
    const int lane = threadIdx.x & 31;
    const long w = (long)blockIdx.x * 8 + (threadIdx.x >> 5); // 0..B*H*T-1
    const int t = (int)(w % T_);
    const int h = (int)((w / T_) % HEADS);
    const int bb = (int)(w / ((long)T_ * HEADS));
    const float* qp = qkv + ((size_t)(bb * T_ + t)) * (3 * DIM) + h * HEAD_DIM;
    const float q0 = qp[lane], q1 = qp[lane + 32];
    const unsigned char* mp = mask + bb * T_;
    float m = -1e30f, l = 0.0f, a0 = 0.0f, a1 = 0.0f;
    for (int j = 0; j < T_; ++j) {
        const float* kp = qkv + ((size_t)(bb * T_ + j)) * (3 * DIM) + DIM + h * HEAD_DIM;
        float s = q0 * kp[lane] + q1 * kp[lane + 32];
#pragma unroll
        for (int off = 16; off > 0; off >>= 1) s += __shfl_xor(s, off, 32);
        s *= 0.125f; // 1/sqrt(64)
        if (!mp[j]) s = -1e30f;
        const float mn = fmaxf(m, s);
        const float corr = __expf(m - mn);
        float p = __expf(s - mn);
        if (s <= -1e29f) p = 0.0f;
        const float* vp = qkv + ((size_t)(bb * T_ + j)) * (3 * DIM) + 2 * DIM + h * HEAD_DIM;
        a0 = a0 * corr + p * vp[lane];
        a1 = a1 * corr + p * vp[lane + 32];
        l = l * corr + p;
        m = mn;
    }
    const float inv = 1.0f / l;
    float* op = out + ((size_t)(bb * T_ + t)) * DIM + h * HEAD_DIM;
    op[lane] = a0 * inv;
    op[lane + 32] = a1 * inv;
}

// ---------------------------------------------------------------------------
extern "C" void kernel_launch(void* const* d_in, const int* in_sizes, int n_in,
                              void* d_out, int out_size, void* d_ws, size_t ws_size,
                              hipStream_t stream)
{
    const float* noisy_mel = (const float*)d_in[0];
    const float* timev     = (const float*)d_in[1];
    const int*   tokens    = (const int*)d_in[2];
    const unsigned char* mask = (const unsigned char*)d_in[3];

    int pi = 4;
    auto NXT = [&]() { return (const float*)d_in[pi++]; };
    const float *in_w = NXT(), *in_b = NXT();
    const float *embed = NXT();
    const float *tproj_w = NXT(), *tproj_b = NXT();
    const float *tconv_w = NXT(), *tconv_b = NXT();
    const float *time_w1 = NXT(), *time_b1 = NXT();
    const float *time_w2 = NXT(), *time_b2 = NXT();
    struct ConvP { const float *dw_w,*dw_b,*ln_g,*ln_b,*pw1_w,*pw1_b,*grn_g,*grn_b,*pw2_w,*pw2_b; } cp[4];
    for (int lyr = 0; lyr < 4; ++lyr)
        cp[lyr] = { NXT(),NXT(),NXT(),NXT(),NXT(),NXT(),NXT(),NXT(),NXT(),NXT() };
    struct DitP { const float *a1w,*a1b,*a2w,*a2b,*qkv_w,*out_w,*ff1_w,*ff1_b,*ff2_w,*ff2_b; } dp[8];
    for (int lyr = 0; lyr < 8; ++lyr)
        dp[lyr] = { NXT(),NXT(),NXT(),NXT(),NXT(),NXT(),NXT(),NXT(),NXT(),NXT() };
    const float *out_g = NXT(), *out_bln = NXT(), *out_w = NXT(), *out_b = NXT();

    // workspace carve-out
    char* ws = (char*)d_ws;
    auto alloc = [&](size_t bytes) { void* r = ws; ws += (bytes + 255) & ~(size_t)255; return r; };
    const size_t MT = (size_t)B_ * T_;
    float* xf    = (float*)alloc(MT * DIM * 4);
    float* x2f   = (float*)alloc(MT * DIM * 4);
    float* af    = (float*)alloc(MT * DIM * 4);
    float* dwf   = (float*)alloc(MT * DIM * 4);
    float* attnf = (float*)alloc(MT * DIM * 4);
    float* hf    = (float*)alloc(MT * FFI * 4);
    float* qf    = (float*)alloc(MT * 3 * DIM * 4);
    float* txtg  = (float*)alloc((size_t)B_ * TTEXT * TEXTDIM * 4);
    float* txtf  = (float*)alloc((size_t)B_ * TTEXT * DIM * 4);
    float* ef    = (float*)alloc((size_t)B_ * DIM * 4);
    float* te1   = (float*)alloc((size_t)B_ * 4 * DIM * 4);
    float* temb  = (float*)alloc((size_t)B_ * DIM * 4);
    float* cond  = (float*)alloc((size_t)B_ * 2 * DIM * 4);
    float* gx    = (float*)alloc((size_t)B_ * FFI * 4);
    float* gm    = (float*)alloc((size_t)B_ * 4);

    auto gemm = [&](const float* A, int lda, const float* Bmat, int ldb,
                    const float* bias, const float* resid,
                    float* C, int ldc, int M, int N, int K, int act) {
        dim3 g((N + 127) / 128, (M + 127) / 128);
        k_gemm_wmma<<<g, 256, 0, stream>>>(A, lda, Bmat, ldb, bias, resid, C, ldc, M, N, K, act);
    };
    auto elems = [](size_t n) { return dim3((unsigned)((n + 255) / 256)); };

    const int M = (int)MT; // 4096 tokens

    // 1) mel input projection: x = mel @ in_w + in_b
    gemm(noisy_mel, NMEL, in_w, DIM, in_b, nullptr, xf, DIM, M, DIM, NMEL, 0);

    // 2) text branch: gather -> proj -> dwconv3 + gelu -> add into x (t < 256)
    k_gather<<<elems((size_t)B_*TTEXT*TEXTDIM), 256, 0, stream>>>(tokens, embed, txtg);
    gemm(txtg, TEXTDIM, tproj_w, DIM, tproj_b, nullptr, txtf, DIM, B_*TTEXT, DIM, TEXTDIM, 0);
    k_txtconv_add<<<elems((size_t)B_*TTEXT*DIM), 256, 0, stream>>>(txtf, tconv_w, tconv_b, xf);

    // 3) time embedding MLP (M=4: skinny GEMMs)
    k_timefreq<<<elems((size_t)B_*DIM), 256, 0, stream>>>(timev, ef);
    k_small_gemm<<<dim3((4*DIM+255)/256, B_), 256, 0, stream>>>(ef, time_w1, time_b1, te1, B_, 4*DIM, DIM, 2);
    k_small_gemm<<<dim3((DIM+255)/256, B_), 256, 0, stream>>>(te1, time_w2, time_b2, temb, B_, DIM, 4*DIM, 0);

    float* cur = xf;
    float* alt = x2f;

    // 4) ConvNeXt blocks
    for (int lyr = 0; lyr < 4; ++lyr) {
        const ConvP& p = cp[lyr];
        k_dwconv7<<<elems(MT * DIM), 256, 0, stream>>>(cur, p.dw_w, p.dw_b, dwf);
        k_ln<<<M, 256, 0, stream>>>(dwf, p.ln_g, p.ln_b, af, DIM);
        gemm(af, DIM, p.pw1_w, FFI, p.pw1_b, nullptr, hf, FFI, M, FFI, DIM, 1); // GELU
        k_grn_gx<<<dim3(FFI/256, B_), 256, 0, stream>>>(hf, gx);
        k_grn_mean<<<B_, 256, 0, stream>>>(gx, gm);
        k_grn_apply<<<elems(MT * FFI), 256, 0, stream>>>(hf, gx, gm, p.grn_g, p.grn_b);
        gemm(hf, FFI, p.pw2_w, DIM, p.pw2_b, cur, alt, DIM, M, DIM, FFI, 0);   // + residual
        float* tswap = cur; cur = alt; alt = tswap;
    }

    // 5) DiT blocks
    for (int lyr = 0; lyr < 8; ++lyr) {
        const DitP& p = dp[lyr];
        // attention half
        k_small_gemm<<<dim3((2*DIM+255)/256, B_), 256, 0, stream>>>(temb, p.a1w, p.a1b, cond, B_, 2*DIM, DIM, 0);
        k_adaln<<<M, 256, 0, stream>>>(cur, cond, af);
        gemm(af, DIM, p.qkv_w, 3*DIM, nullptr, nullptr, qf, 3*DIM, M, 3*DIM, DIM, 0);
        k_rope<<<elems((size_t)B_*T_*32), 256, 0, stream>>>(qf);
        k_attn<<<(B_*HEADS*T_)/8, 256, 0, stream>>>(qf, mask, attnf);
        gemm(attnf, DIM, p.out_w, DIM, nullptr, cur, alt, DIM, M, DIM, DIM, 0); // + residual
        { float* tswap = cur; cur = alt; alt = tswap; }
        // FF half
        k_small_gemm<<<dim3((2*DIM+255)/256, B_), 256, 0, stream>>>(temb, p.a2w, p.a2b, cond, B_, 2*DIM, DIM, 0);
        k_adaln<<<M, 256, 0, stream>>>(cur, cond, af);
        gemm(af, DIM, p.ff1_w, FFI, p.ff1_b, nullptr, hf, FFI, M, FFI, DIM, 1); // GELU
        gemm(hf, FFI, p.ff2_w, DIM, p.ff2_b, cur, alt, DIM, M, DIM, FFI, 0);    // + residual
        { float* tswap = cur; cur = alt; alt = tswap; }
    }

    // 6) final LN + output projection -> d_out [B,T,NMEL] fp32
    k_ln<<<M, 256, 0, stream>>>(cur, out_g, out_bln, af, DIM);
    gemm(af, DIM, out_w, NMEL, out_b, nullptr, (float*)d_out, NMEL, M, NMEL, DIM, 0);

    (void)n_in; (void)in_sizes; (void)out_size; (void)ws_size;
}
